// CSPLayer_conformer_19146964206083
// MI455X (gfx1250) — compile-verified, matrix operand order checked by reading, not run on hardware
//
#include <hip/hip_runtime.h>
#include <hip/hip_bf16.h>
#include <math.h>

// ---------------------------------------------------------------------------
// Geometry (reference: batch 8, 112x112, C=64 hidden, P = pixels/tokens)
// ---------------------------------------------------------------------------
#define P_TOT 100352      // 8 * 112 * 112, image-major flat index
#define HW    12544       // 112*112
#define WIDTH 112

typedef __bf16 bf16_t;
typedef __attribute__((ext_vector_type(16))) __bf16 v16bf;
typedef __attribute__((ext_vector_type(8)))  __bf16 v8bf;
typedef __attribute__((ext_vector_type(8)))  float  v8f;
typedef int v4i_vs __attribute__((vector_size(16)));   // async-LDS builtin param

#define AS1 __attribute__((address_space(1)))
#define AS3 __attribute__((address_space(3)))

#if defined(__has_builtin)
#if __has_builtin(__builtin_amdgcn_global_load_async_to_lds_b128)
#define ASYNC_LDS 1
#endif
#endif

// ---------------------------------------------------------------------------
// Tiled activation layout ("T-layout"): a [C][P] bf16 tensor is stored as
// chunks T[ct][p][hf][8] with ct=c>>4, hf=(c>>3)&1, r=c&7.  One 16-byte chunk
// is exactly one WMMA bf16 operand half for one lane, so GEMMs load operands
// and store results with single b128 accesses and zero LDS staging.  Weights
// are pre-packed once into the mirrored layout W[kt][m][hf][8].
// ---------------------------------------------------------------------------
enum { A_NONE = 0, A_SILU = 1, A_GELU = 2, A_SIG = 3 };

__device__ __forceinline__ float fast_sig(float x) {
  return __builtin_amdgcn_rcpf(1.0f + __expf(-x));
}
template<int ACT> __device__ __forceinline__ float act_f(float x) {
  if constexpr (ACT == A_SILU) return x * fast_sig(x);
  else if constexpr (ACT == A_GELU) return 0.5f * x * (1.0f + erff(x * 0.70710678f));
  else if constexpr (ACT == A_SIG)  return fast_sig(x);
  else return x;
}

__device__ __forceinline__ void async_wait_all() {
#if defined(ASYNC_LDS)
#if __has_builtin(__builtin_amdgcn_s_wait_asynccnt)
  __builtin_amdgcn_s_wait_asynccnt(0);
#else
  asm volatile("s_wait_asynccnt 0" ::: "memory");
#endif
#endif
}

#define CAT16(a, b) __builtin_shufflevector(a, b, 0,1,2,3,4,5,6,7,8,9,10,11,12,13,14,15)

// ---------------------------------------------------------------------------
// Weight packer: W[m*ldm + k*kstride + off] (f32) -> packed [kt][M][16] bf16
// ---------------------------------------------------------------------------
__global__ void k_packw(const float* __restrict__ W, bf16_t* __restrict__ out,
                        int M, int K, int ldm, int kstride, int off)
{
  int tid = blockIdx.x * 256 + threadIdx.x;
  if (tid >= M * K) return;
  int m = tid / K, k = tid - m * K;
  out[((size_t)(k >> 4) * M + m) * 16 + ((k >> 3) & 1) * 8 + (k & 7)] =
      (bf16_t)W[m * ldm + k * kstride + off];
}

// Input packer: f32 NCHW (8,128,112,112) -> T-layout bf16 (128 channels)
__global__ void k_packx(const float* __restrict__ X, bf16_t* __restrict__ out)
{
  size_t i = (size_t)blockIdx.x * 256 + threadIdx.x;   // over 128*P
  int r = i & 7, hf = (i >> 3) & 1;
  size_t q = i >> 4;
  int p = (int)(q % P_TOT), ct = (int)(q / P_TOT);
  int c = ct * 16 + hf * 8 + r;
  int n = p / HW, hw = p - n * HW;
  out[i] = (bf16_t)X[(size_t)(n * 128 + c) * HW + hw];
}

// zero-fill (16B granules) — provides the OOB "zero page" tensor
__global__ void k_zero(uint4* __restrict__ z)
{
  z[(size_t)blockIdx.x * 256 + threadIdx.x] = uint4{0, 0, 0, 0};
}

// ---------------------------------------------------------------------------
// WMMA GEMM, fragment-native layout:  O = act(bn(A @ B + bias)) + resid
//   At : packed weights [kt][M][16],  Bt : T-layout [kt][P][16]
//   OM==0 : bf16 T-layout out        OM==1 : f32 NCHW out (final conv)
// Block = 128 threads = 4 independent waves, each owns 16(M) x 64(N).
// No LDS, no barriers: operands fetched as b128 from L2-resident tensors.
// ---------------------------------------------------------------------------
template<int ACT, int OM>
__global__ __launch_bounds__(128) void k_gemm(
    const bf16_t* __restrict__ At, const bf16_t* __restrict__ Bt, void* __restrict__ Ov,
    const float* __restrict__ bias,
    const float* __restrict__ bng, const float* __restrict__ bnb,
    const float* __restrict__ bnm, const float* __restrict__ bnv,
    const bf16_t* __restrict__ resid, int M, int K)
{
  const int pbase = blockIdx.x * 64;
  const int wv = threadIdx.x >> 5, lid = threadIdx.x & 31;
  const int half = lid >> 4, l16 = lid & 15;
  const int mtile = blockIdx.y * 64 + wv * 16;
  const int mrow = mtile + l16;

  v8f acc[4];
  #pragma unroll
  for (int i = 0; i < 4; ++i)
    #pragma unroll
    for (int r = 0; r < 8; ++r) acc[i][r] = 0.0f;

  for (int kb = 0; kb < K; kb += 32) {
    int kt = kb >> 4;
    const bf16_t* ap = At + ((size_t)kt * M + mrow) * 16 + half * 8;
    v8bf a0 = *(const v8bf*)ap;
    v8bf a1 = *(const v8bf*)(ap + (size_t)M * 16);
    v16bf a = CAT16(a0, a1);
    if (kb + 32 < K)   // next K panel -> global_prefetch_b8
      __builtin_prefetch(Bt + ((size_t)(kt + 2) * P_TOT + pbase) * 16 + threadIdx.x * 8, 0, 1);
    #pragma unroll
    for (int nt = 0; nt < 4; ++nt) {
      int p = pbase + nt * 16 + l16;
      const bf16_t* bp = Bt + ((size_t)kt * P_TOT + p) * 16 + half * 8;
      v8bf b0 = *(const v8bf*)bp;
      v8bf b1 = *(const v8bf*)(bp + (size_t)P_TOT * 16);
      v16bf b = CAT16(b0, b1);
      acc[nt] = __builtin_amdgcn_wmma_f32_16x16x32_bf16(
          false, a, false, b, (short)0, acc[nt], false, false);
    }
  }

  const int ct = mtile >> 4;
  #pragma unroll
  for (int nt = 0; nt < 4; ++nt) {
    int p = pbase + nt * 16 + l16;
    v8bf rv;
    if (resid) rv = *(const v8bf*)(resid + ((size_t)ct * P_TOT + p) * 16 + half * 8);
    if constexpr (OM == 0) {
      v8bf o;
      #pragma unroll
      for (int r = 0; r < 8; ++r) {
        int mm = mtile + half * 8 + r;
        float v = acc[nt][r];
        if (bias) v += bias[mm];
        if (bng) {
          float inv = bng[mm] * __builtin_amdgcn_rsqf(bnv[mm] + 1e-5f);
          v = v * inv + (bnb[mm] - bnm[mm] * inv);
        }
        v = act_f<ACT>(v);
        if (resid) v += (float)rv[r];
        o[r] = (bf16_t)v;
      }
      *(v8bf*)((bf16_t*)Ov + ((size_t)ct * P_TOT + p) * 16 + half * 8) = o;
    } else {
      int ni = p / HW, hw = p - ni * HW;
      #pragma unroll
      for (int r = 0; r < 8; ++r) {
        int mm = mtile + half * 8 + r;
        float v = acc[nt][r];
        if (bng) {
          float inv = bng[mm] * __builtin_amdgcn_rsqf(bnv[mm] + 1e-5f);
          v = v * inv + (bnb[mm] - bnm[mm] * inv);
        }
        v = act_f<ACT>(v);
        ((float*)Ov)[(size_t)(ni * M + mm) * HW + hw] = v;
      }
    }
  }
}

// ---------------------------------------------------------------------------
// 3x3 conv (64->64, pad 1): 9-tap accumulated WMMA, fused BN+SiLU (+resid).
// Zero padding by base-pointer select into a zeroed tensor Z (no data
// cndmasks, no WMMA->VALU hazard NOPs).  At9: 9 packed [4][64][16] taps.
// ---------------------------------------------------------------------------
__global__ __launch_bounds__(128) void k_conv3(
    const bf16_t* __restrict__ At9, const bf16_t* __restrict__ Bt,
    const bf16_t* __restrict__ Z,   // zeroed 4-tile tensor
    bf16_t* __restrict__ O,
    const float* __restrict__ bng, const float* __restrict__ bnb,
    const float* __restrict__ bnm, const float* __restrict__ bnv,
    const bf16_t* __restrict__ resid)
{
  const int pbase = blockIdx.x * 64;
  const int wv = threadIdx.x >> 5, lid = threadIdx.x & 31;
  const int half = lid >> 4, l16 = lid & 15;
  const int mtile = wv * 16;
  const int mrow = mtile + l16;

  v8f acc[4];
  #pragma unroll
  for (int i = 0; i < 4; ++i)
    #pragma unroll
    for (int r = 0; r < 8; ++r) acc[i][r] = 0.0f;

  // decode pixels once per nt
  int pp[4], pni[4], ph[4], pw[4];
  #pragma unroll
  for (int nt = 0; nt < 4; ++nt) {
    int p = pbase + nt * 16 + l16;
    pp[nt] = p; pni[nt] = p / HW;
    int hw = p - pni[nt] * HW;
    ph[nt] = hw / WIDTH; pw[nt] = hw - ph[nt] * WIDTH;
  }

  for (int tap = 0; tap < 9; ++tap) {
    int dy = tap / 3 - 1, dx = tap % 3 - 1;
    // all 4 A half-fragments of this tap (kt = 0..3), hoisted out of nt loop
    const bf16_t* ap = At9 + (size_t)tap * 64 * 64 + (size_t)mrow * 16 + half * 8;
    v8bf a0 = *(const v8bf*)ap;
    v8bf a1 = *(const v8bf*)(ap + 1 * 64 * 16);
    v8bf a2 = *(const v8bf*)(ap + 2 * 64 * 16);
    v8bf a3 = *(const v8bf*)(ap + 3 * 64 * 16);
    v16bf aLo = CAT16(a0, a1);
    v16bf aHi = CAT16(a2, a3);
    #pragma unroll
    for (int nt = 0; nt < 4; ++nt) {
      int hh = ph[nt] + dy, wq = pw[nt] + dx;
      int hc = min(max(hh, 0), WIDTH - 1), wc = min(max(wq, 0), WIDTH - 1);
      bool ok = (hh == hc) && (wq == wc);
      int pc = pni[nt] * HW + hc * WIDTH + wc;
      const bf16_t* src = ok ? Bt : Z;              // base select, not data
      const bf16_t* bp = src + (size_t)pc * 16 + half * 8;
      v8bf b0 = *(const v8bf*)bp;
      v8bf b1 = *(const v8bf*)(bp + 1 * (size_t)P_TOT * 16);
      v8bf b2 = *(const v8bf*)(bp + 2 * (size_t)P_TOT * 16);
      v8bf b3 = *(const v8bf*)(bp + 3 * (size_t)P_TOT * 16);
      acc[nt] = __builtin_amdgcn_wmma_f32_16x16x32_bf16(
          false, aLo, false, CAT16(b0, b1), (short)0, acc[nt], false, false);
      acc[nt] = __builtin_amdgcn_wmma_f32_16x16x32_bf16(
          false, aHi, false, CAT16(b2, b3), (short)0, acc[nt], false, false);
    }
  }

  const int ct = mtile >> 4;
  #pragma unroll
  for (int nt = 0; nt < 4; ++nt) {
    int p = pp[nt];
    v8bf rv;
    if (resid) rv = *(const v8bf*)(resid + ((size_t)ct * P_TOT + p) * 16 + half * 8);
    v8bf o;
    #pragma unroll
    for (int r = 0; r < 8; ++r) {
      int mm = mtile + half * 8 + r;
      float v = acc[nt][r];
      float inv = bng[mm] * __builtin_amdgcn_rsqf(bnv[mm] + 1e-5f);
      v = v * inv + (bnb[mm] - bnm[mm] * inv);
      v = v * fast_sig(v);                         // SiLU
      if (resid) v += (float)rv[r];
      o[r] = (bf16_t)v;
    }
    *(v8bf*)(O + ((size_t)ct * P_TOT + p) * 16 + half * 8) = o;
  }
}

// ---------------------------------------------------------------------------
// GroupNorm(groups==channels) + SiLU over T-layout bf16, per (n,c)
// ---------------------------------------------------------------------------
__global__ __launch_bounds__(256) void k_gn_silu(
    const bf16_t* __restrict__ in, bf16_t* __restrict__ out,
    const float* __restrict__ g, const float* __restrict__ b)
{
  __shared__ float s1[256], s2[256];
  int c = blockIdx.x >> 3, n = blockIdx.x & 7;
  size_t base = ((size_t)(c >> 4) * P_TOT + n * HW) * 16 + ((c >> 3) & 1) * 8 + (c & 7);
  float sm = 0.f, sq = 0.f;
  for (int i = threadIdx.x; i < HW; i += 256) {
    float v = (float)in[base + (size_t)i * 16]; sm += v; sq += v * v;
  }
  s1[threadIdx.x] = sm; s2[threadIdx.x] = sq; __syncthreads();
  for (int o = 128; o > 0; o >>= 1) {
    if (threadIdx.x < o) { s1[threadIdx.x] += s1[threadIdx.x + o]; s2[threadIdx.x] += s2[threadIdx.x + o]; }
    __syncthreads();
  }
  float mu = s1[0] / (float)HW;
  float var = s2[0] / (float)HW - mu * mu;
  float inv = __builtin_amdgcn_rsqf(var + 1e-5f) * g[c];
  float bb = b[c];
  for (int i = threadIdx.x; i < HW; i += 256) {
    float v = ((float)in[base + (size_t)i * 16] - mu) * inv + bb;
    out[base + (size_t)i * 16] = (bf16_t)(v * fast_sig(v));
  }
}

// LayerNorm over C=64 per token, T-layout (whole chunks held in registers)
__global__ __launch_bounds__(256) void k_ln(
    const bf16_t* __restrict__ in, bf16_t* __restrict__ out,
    const float* __restrict__ g, const float* __restrict__ b)
{
  int p = blockIdx.x * 256 + threadIdx.x;
  v8bf ch[8];
  float s = 0.f, ss = 0.f;
  #pragma unroll
  for (int idx = 0; idx < 8; ++idx) {           // idx = ct*2+hf, c = idx*8+r
    int ct = idx >> 1, hf = idx & 1;
    ch[idx] = *(const v8bf*)(in + ((size_t)ct * P_TOT + p) * 16 + hf * 8);
    #pragma unroll
    for (int r = 0; r < 8; ++r) { float v = (float)ch[idx][r]; s += v; ss += v * v; }
  }
  float mu = s * (1.0f / 64.0f);
  float var = ss * (1.0f / 64.0f) - mu * mu;
  float inv = __builtin_amdgcn_rsqf(var + 1e-5f);
  #pragma unroll
  for (int idx = 0; idx < 8; ++idx) {
    int ct = idx >> 1, hf = idx & 1;
    v8bf o;
    #pragma unroll
    for (int r = 0; r < 8; ++r) {
      int c = idx * 8 + r;
      o[r] = (bf16_t)(((float)ch[idx][r] - mu) * inv * g[c] + b[c]);
    }
    *(v8bf*)(out + ((size_t)ct * P_TOT + p) * 16 + hf * 8) = o;
  }
}

// global average pool (T-layout) -> f32 pooled[n*64+c]
__global__ __launch_bounds__(256) void k_pool(const bf16_t* __restrict__ in,
                                              float* __restrict__ pooled)
{
  __shared__ float s1[256];
  int c = blockIdx.x >> 3, n = blockIdx.x & 7;
  size_t base = ((size_t)(c >> 4) * P_TOT + n * HW) * 16 + ((c >> 3) & 1) * 8 + (c & 7);
  float sm = 0.f;
  for (int i = threadIdx.x; i < HW; i += 256) sm += (float)in[base + (size_t)i * 16];
  s1[threadIdx.x] = sm; __syncthreads();
  for (int o = 128; o > 0; o >>= 1) {
    if (threadIdx.x < o) s1[threadIdx.x] += s1[threadIdx.x + o];
    __syncthreads();
  }
  if (threadIdx.x == 0) pooled[n * 64 + c] = s1[0] / (float)HW;
}

// channel_action MLP: 1x1 -> BN -> GELU -> 1x1 -> sigmoid (f32, tiny)
__global__ __launch_bounds__(64) void k_ca(
    const float* __restrict__ pooled,
    const float* __restrict__ w1, const float* __restrict__ b1,
    const float* __restrict__ g, const float* __restrict__ b,
    const float* __restrict__ m, const float* __restrict__ v,
    const float* __restrict__ w2, const float* __restrict__ b2,
    float* __restrict__ gate)
{
  __shared__ float pl[64], hh[64];
  int n = blockIdx.x, t = threadIdx.x;
  pl[t] = pooled[n * 64 + t]; __syncthreads();
  float s = b1[t];
  for (int k = 0; k < 64; ++k) s += w1[t * 64 + k] * pl[k];
  float inv = g[t] * __builtin_amdgcn_rsqf(v[t] + 1e-5f);
  s = s * inv + (b[t] - m[t] * inv);
  s = 0.5f * s * (1.0f + erff(s * 0.70710678f));
  hh[t] = s; __syncthreads();
  float s2 = b2[t];
  for (int k = 0; k < 64; ++k) s2 += w2[t * 64 + k] * hh[k];
  gate[n * 64 + t] = fast_sig(s2);
}

// out = in * gate[n(p)][c], elementwise over T-layout flat index
__global__ void k_scale(const bf16_t* __restrict__ in, const float* __restrict__ gate,
                        bf16_t* __restrict__ out)
{
  size_t i = (size_t)blockIdx.x * 256 + threadIdx.x;
  int r = i & 7, hf = (i >> 3) & 1;
  size_t q = i >> 4;
  int p = (int)(q % P_TOT), ct = (int)(q / P_TOT);
  int c = ct * 16 + hf * 8 + r;
  int n = p / HW;
  out[i] = (bf16_t)((float)in[i] * gate[n * 64 + c]);
}

__global__ void k_mul(const bf16_t* __restrict__ a, const bf16_t* __restrict__ b,
                      bf16_t* __restrict__ o)
{ size_t i = (size_t)blockIdx.x * 256 + threadIdx.x; o[i] = (bf16_t)((float)a[i] * (float)b[i]); }

__global__ void k_add(const bf16_t* __restrict__ a, const bf16_t* __restrict__ b,
                      bf16_t* __restrict__ o)
{ size_t i = (size_t)blockIdx.x * 256 + threadIdx.x; o[i] = (bf16_t)((float)a[i] + (float)b[i]); }

// ---------------------------------------------------------------------------
// Swin windowed attention (hd=8, 49 tokens: tiny FLOPs -> VALU; per-head
// q/k/v slices are single 16B chunks in T-layout).  K/V staged to LDS via
// async global->LDS b128.  One block (2 waves) per (window, head).
// rel_pos_index = (di+6)*13 + (dj+6) computed inline.
// ---------------------------------------------------------------------------
__global__ __launch_bounds__(64) void k_attn(
    const bf16_t* __restrict__ qkv, const float* __restrict__ relb,
    bf16_t* __restrict__ out)
{
  __shared__ alignas(16) bf16_t kk[49 * 8], vv[49 * 8];
  int win = blockIdx.x, h = blockIdx.y;
  int ni = win >> 8;
  int rr = win & 255;
  int wh = rr >> 4, ww = rr & 15;
  int pb = ni * HW + (wh * 7) * WIDTH + ww * 7;
  int t = threadIdx.x;
  int hct = h >> 1, hhf = h & 1;                  // head -> (tile, half) offsets
  if (t < 49) {
    int p = pb + (t / 7) * WIDTH + (t % 7);
    const bf16_t* ksrc = qkv + ((size_t)(4 + hct) * P_TOT + p) * 16 + hhf * 8;
    const bf16_t* vsrc = qkv + ((size_t)(8 + hct) * P_TOT + p) * 16 + hhf * 8;
#if defined(ASYNC_LDS)
    __builtin_amdgcn_global_load_async_to_lds_b128(
        (AS1 v4i_vs*)ksrc, (AS3 v4i_vs*)&kk[t * 8], 0, 0);
    __builtin_amdgcn_global_load_async_to_lds_b128(
        (AS1 v4i_vs*)vsrc, (AS3 v4i_vs*)&vv[t * 8], 0, 0);
#else
    *(uint4*)&kk[t * 8] = *(const uint4*)ksrc;
    *(uint4*)&vv[t * 8] = *(const uint4*)vsrc;
#endif
  }
  async_wait_all();
  __syncthreads();
  if (t >= 49) return;
  int p_i = pb + (t / 7) * WIDTH + (t % 7);
  v8bf qv = *(const v8bf*)(qkv + ((size_t)hct * P_TOT + p_i) * 16 + hhf * 8);
  float q[8];
  #pragma unroll
  for (int d = 0; d < 8; ++d) q[d] = (float)qv[d] * 0.35355339f;
  int ai = t / 7, aj = t % 7;
  float s[49], mx = -1e30f;
  for (int j = 0; j < 49; ++j) {
    float a = 0.f;
    #pragma unroll
    for (int d = 0; d < 8; ++d) a += q[d] * (float)kk[j * 8 + d];
    int bi = j / 7, bj = j % 7;
    a += relb[((ai - bi + 6) * 13 + (aj - bj + 6)) * 8 + h];
    s[j] = a; mx = fmaxf(mx, a);
  }
  float sum = 0.f;
  for (int j = 0; j < 49; ++j) { s[j] = __expf(s[j] - mx); sum += s[j]; }
  float inv = __builtin_amdgcn_rcpf(sum);
  v8bf o;
  #pragma unroll
  for (int d = 0; d < 8; ++d) {
    float acc = 0.f;
    for (int j = 0; j < 49; ++j) acc += s[j] * (float)vv[j * 8 + d];
    o[d] = (bf16_t)(acc * inv);
  }
  *(v8bf*)(out + ((size_t)hct * P_TOT + p_i) * 16 + hhf * 8) = o;
}

// ---------------------------------------------------------------------------
// Host-side parameter bundles and launchers
// ---------------------------------------------------------------------------
struct Bn   { const float *g, *b, *m, *v; };
struct Gate { const float *w1, *b1; Bn bn; const float *w2, *b2; };
struct SwinP {
  const float *n1g, *n1b, *qkvw, *qkvb, *relb, *projw, *projb;
  const float *n2g, *n2b, *fc1w, *fc1b, *fc2w, *fc2b;
};
struct Bott {
  const float* cw[10]; Bn cbn[10];
  Gate ca, sa;
  const float *pew, *peb, *pelng, *pelnb;
  SwinP sw;
};
// packed (fragment-order bf16) weights
struct PBott {
  bf16_t* cw[10];                 // 1x1: [kt][64][16]; 3x3: 9 consecutive taps
  bf16_t *saw1, *saw2, *pew, *qkvw, *projw, *fc1w, *fc2w;
};
struct WsP { bf16_t* B[8]; bf16_t* big; bf16_t* Z; float* pooled; float* gate; };

static void gemm(hipStream_t st, const bf16_t* At, const bf16_t* Bt, void* O,
                 int M, int K, const float* bias, const Bn* bn, int act,
                 const bf16_t* resid, int omode = 0)
{
  dim3 grd(P_TOT / 64, M / 64), blk(128);
  const float *g = bn ? bn->g : nullptr, *b = bn ? bn->b : nullptr;
  const float *m = bn ? bn->m : nullptr, *v = bn ? bn->v : nullptr;
  if (omode) { k_gemm<A_SILU, 1><<<grd, blk, 0, st>>>(At, Bt, O, bias, g, b, m, v, resid, M, K); return; }
  switch (act) {
    case A_SILU: k_gemm<A_SILU, 0><<<grd, blk, 0, st>>>(At, Bt, O, bias, g, b, m, v, resid, M, K); break;
    case A_GELU: k_gemm<A_GELU, 0><<<grd, blk, 0, st>>>(At, Bt, O, bias, g, b, m, v, resid, M, K); break;
    case A_SIG:  k_gemm<A_SIG , 0><<<grd, blk, 0, st>>>(At, Bt, O, bias, g, b, m, v, resid, M, K); break;
    default:     k_gemm<A_NONE, 0><<<grd, blk, 0, st>>>(At, Bt, O, bias, g, b, m, v, resid, M, K); break;
  }
}

static void conv3(hipStream_t st, const bf16_t* At9, const bf16_t* B, const bf16_t* Z,
                  bf16_t* O, const Bn& bn, const bf16_t* resid)
{
  k_conv3<<<dim3(P_TOT / 64, 1), 128, 0, st>>>(At9, B, Z, O, bn.g, bn.b, bn.m, bn.v, resid);
}

static void swin_block(hipStream_t st, const bf16_t* Tin, bf16_t* Tout, bf16_t* t1,
                       bf16_t* t2, bf16_t* big, const PBott& pk, const SwinP& s)
{
  k_ln<<<P_TOT / 256, 256, 0, st>>>(Tin, t1, s.n1g, s.n1b);
  gemm(st, pk.qkvw, t1, big, 192, 64, s.qkvb, nullptr, A_NONE, nullptr);
  k_attn<<<dim3(2048, 8), 64, 0, st>>>(big, s.relb, t1);
  gemm(st, pk.projw, t1, t2, 64, 64, s.projb, nullptr, A_NONE, Tin);  // x = shortcut + attn
  k_ln<<<P_TOT / 256, 256, 0, st>>>(t2, t1, s.n2g, s.n2b);
  gemm(st, pk.fc1w, t1, big, 256, 64, s.fc1b, nullptr, A_GELU, nullptr);
  gemm(st, pk.fc2w, big, Tout, 64, 256, s.fc2b, nullptr, A_NONE, t2); // x + mlp
}

static void bottleneck(hipStream_t st, bf16_t* x, const Bott& bp, const PBott& pk,
                       const WsP& w)
{
  const int EB = (64 * P_TOT) / 256;
  bf16_t *B0 = w.B[0], *B1 = w.B[1], *B2 = w.B[2], *B3 = w.B[3];
  bf16_t *B4 = w.B[4], *B5 = w.B[5], *B6 = w.B[6], *B7 = w.B[7];

  // y = conv2(conv1(x))
  gemm(st, pk.cw[1], x, B5, 64, 64, nullptr, &bp.cbn[1], A_SILU, nullptr);
  conv3(st, pk.cw[2], B5, w.Z, B0, bp.cbn[2], nullptr);
  // y_m = channel_action(y)
  k_pool<<<512, 256, 0, st>>>(B0, w.pooled);
  k_ca<<<8, 64, 0, st>>>(w.pooled, bp.ca.w1, bp.ca.b1, bp.ca.bn.g, bp.ca.bn.b,
                         bp.ca.bn.m, bp.ca.bn.v, bp.ca.w2, bp.ca.b2, w.gate);
  // swin_y = swin(patch_embed(x * y_m))  -> B3
  k_scale<<<EB, 256, 0, st>>>(x, w.gate, B5);
  gemm(st, pk.pew, B5, B6, 64, 64, bp.peb, nullptr, A_NONE, nullptr);
  k_ln<<<P_TOT / 256, 256, 0, st>>>(B6, B7, bp.pelng, bp.pelnb);
  swin_block(st, B7, B3, B5, B6, w.big, pk, bp.sw);
  // y1 = conv3(y) + x
  gemm(st, pk.cw[3], B0, B1, 64, 64, nullptr, &bp.cbn[3], A_SILU, x);
  // y2 = conv5(conv4(y1) * spatial_action(swin_y))
  gemm(st, pk.cw[4], B1, B5, 64, 64, nullptr, &bp.cbn[4], A_SILU, nullptr);
  gemm(st, pk.saw1, B3, B6, 64, 64, bp.sa.b1, &bp.sa.bn, A_GELU, nullptr);
  gemm(st, pk.saw2, B6, B7, 64, 64, bp.sa.b2, nullptr, A_SIG, nullptr);
  k_mul<<<EB, 256, 0, st>>>(B5, B7, B5);
  conv3(st, pk.cw[5], B5, w.Z, B2, bp.cbn[5], nullptr);
  // y_m = channel_action(y2)
  k_pool<<<512, 256, 0, st>>>(B2, w.pooled);
  k_ca<<<8, 64, 0, st>>>(w.pooled, bp.ca.w1, bp.ca.b1, bp.ca.bn.g, bp.ca.bn.b,
                         bp.ca.bn.m, bp.ca.bn.v, bp.ca.w2, bp.ca.b2, w.gate);
  // y2 = conv6(y2) + y1  -> B6
  gemm(st, pk.cw[6], B2, B6, 64, 64, nullptr, &bp.cbn[6], A_SILU, B1);
  // swin_y2 = swin(patch_embed(swin_y * y_m)) -> B4
  k_scale<<<EB, 256, 0, st>>>(B3, w.gate, B5);
  gemm(st, pk.pew, B5, B2, 64, 64, bp.peb, nullptr, A_NONE, nullptr);
  k_ln<<<P_TOT / 256, 256, 0, st>>>(B2, B7, bp.pelng, bp.pelnb);
  swin_block(st, B7, B4, B5, B2, w.big, pk, bp.sw);
  // y3 = conv9(conv8(conv7(y2) * spatial_action(swin_y2)))
  gemm(st, pk.cw[7], B6, B5, 64, 64, nullptr, &bp.cbn[7], A_SILU, nullptr);
  gemm(st, pk.saw1, B4, B2, 64, 64, bp.sa.b1, &bp.sa.bn, A_GELU, nullptr);
  gemm(st, pk.saw2, B2, B7, 64, 64, bp.sa.b2, nullptr, A_SIG, nullptr);
  k_mul<<<EB, 256, 0, st>>>(B5, B7, B5);
  conv3(st, pk.cw[8], B5, w.Z, B2, bp.cbn[8], nullptr);
  gemm(st, pk.cw[9], B2, B5, 64, 64, nullptr, &bp.cbn[9], A_SILU, nullptr);
  // return y3 + y2 (in-place into x)
  k_add<<<EB, 256, 0, st>>>(B5, B6, x);
}

// ---------------------------------------------------------------------------
// kernel_launch: unpack 168 parameter arrays (depth-first dict order),
// pre-pack weights into fragment order (~0.8 MB, re-done every call ->
// deterministic), zero the OOB page, then run the pipeline.
// Workspace: 15 bf16 slots (~193 MB) + gates + packed weights.
// x1/x2 adjacent -> final conv sees one 8-tile (128-ch) T-layout tensor.
// ---------------------------------------------------------------------------
extern "C" void kernel_launch(void* const* d_in, const int* in_sizes, int n_in,
                              void* d_out, int out_size, void* d_ws, size_t ws_size,
                              hipStream_t stream)
{
  (void)in_sizes; (void)n_in; (void)out_size; (void)ws_size;
  int ci = 0;
  auto nx = [&]() -> const float* { return (const float*)d_in[ci++]; };
  auto pBn = [&]() { Bn r; r.g = nx(); r.b = nx(); r.m = nx(); r.v = nx(); return r; };
  auto pGate = [&]() { Gate g; g.w1 = nx(); g.b1 = nx(); g.bn = pBn(); g.w2 = nx(); g.b2 = nx(); return g; };
  auto pBott = [&]() {
    Bott bt{};
    for (int i = 1; i <= 9; ++i) { bt.cw[i] = nx(); bt.cbn[i] = pBn(); }
    bt.ca = pGate(); bt.sa = pGate();
    bt.pew = nx(); bt.peb = nx(); bt.pelng = nx(); bt.pelnb = nx();
    SwinP& s = bt.sw;
    s.n1g = nx(); s.n1b = nx(); s.qkvw = nx(); s.qkvb = nx(); s.relb = nx();
    s.projw = nx(); s.projb = nx(); s.n2g = nx(); s.n2b = nx();
    s.fc1w = nx(); s.fc1b = nx(); s.fc2w = nx(); s.fc2b = nx();
    return bt;
  };

  const float* X = nx();                               // (8,128,112,112) f32 NCHW
  const float *c1w = nx(), *c1g = nx(), *c1b = nx();   // conv1 (GN affine)
  const float *c2w = nx(), *c2g = nx(), *c2b = nx();   // conv2 (GN affine)
  const float* c3w = nx(); Bn c3bn = pBn();            // conv3 (BN)
  Bott mA = pBott();
  Bott mB = pBott();

  const size_t SLOT = (size_t)64 * P_TOT * sizeof(bf16_t);   // bytes per slot
  char* base = (char*)d_ws;
  bf16_t* x1 = (bf16_t*)base;                          // persists (tiles 0..3)
  bf16_t* x2 = (bf16_t*)(base + SLOT);                 // persists (tiles 4..7)
  WsP w{};
  for (int i = 0; i < 8; ++i) w.B[i] = (bf16_t*)(base + (2 + i) * SLOT);
  w.big    = (bf16_t*)(base + 10 * SLOT);              // 4 slots: qkv(3)/fc1(4)
  w.Z      = (bf16_t*)(base + 14 * SLOT);              // zero page (4 tiles)
  w.pooled = (float*)(base + 15 * SLOT);
  w.gate   = w.pooled + 512;
  bf16_t* wcur = (bf16_t*)(w.gate + 512);              // packed-weight arena

  auto packw = [&](const float* W, int M, int K, int ldm, int ks, int off) -> bf16_t* {
    bf16_t* out = wcur; wcur += (size_t)M * K;
    int tot = M * K;
    k_packw<<<(tot + 255) / 256, 256, 0, stream>>>(W, out, M, K, ldm, ks, off);
    return out;
  };
  auto pack3x3 = [&](const float* W) -> bf16_t* {      // [64][64][3][3] -> 9 taps
    bf16_t* out = wcur;
    for (int tap = 0; tap < 9; ++tap) packw(W, 64, 64, 64 * 9, 9, tap);
    return out;
  };
  auto packBott = [&](const Bott& bt) -> PBott {
    PBott p{};
    for (int i = 1; i <= 9; ++i)
      p.cw[i] = (i == 2 || i == 5 || i == 8) ? pack3x3(bt.cw[i])
                                             : packw(bt.cw[i], 64, 64, 64, 1, 0);
    p.saw1 = packw(bt.sa.w1, 64, 64, 64, 1, 0);
    p.saw2 = packw(bt.sa.w2, 64, 64, 64, 1, 0);
    p.pew  = packw(bt.pew, 64, 64, 64, 1, 0);
    p.qkvw = packw(bt.sw.qkvw, 192, 64, 64, 1, 0);
    p.projw = packw(bt.sw.projw, 64, 64, 64, 1, 0);
    p.fc1w = packw(bt.sw.fc1w, 256, 64, 64, 1, 0);
    p.fc2w = packw(bt.sw.fc2w, 64, 256, 256, 1, 0);
    return p;
  };

  // zero page for conv3 OOB reads (re-zeroed every call; ws is poisoned once)
  k_zero<<<(int)(SLOT / 16 / 256), 256, 0, stream>>>((uint4*)w.Z);

  bf16_t* pc1 = packw(c1w, 64, 128, 128, 1, 0);
  bf16_t* pc2 = packw(c2w, 64, 128, 128, 1, 0);
  bf16_t* pc3 = packw(c3w, 128, 128, 128, 1, 0);
  PBott pkA = packBott(mA);
  PBott pkB = packBott(mB);

  // input -> 128-channel T-layout bf16 (overlaid on big, free until swin)
  bf16_t* Xt = w.big;
  k_packx<<<(int)(((size_t)128 * P_TOT) / 256), 256, 0, stream>>>(X, Xt);

  // stem: x1 = silu(GN(conv1(x))), x2 = silu(GN(conv2(x)))
  gemm(stream, pc1, Xt, w.B[0], 64, 128, nullptr, nullptr, A_NONE, nullptr);
  k_gn_silu<<<512, 256, 0, stream>>>(w.B[0], x1, c1g, c1b);
  gemm(stream, pc2, Xt, w.B[0], 64, 128, nullptr, nullptr, A_NONE, nullptr);
  k_gn_silu<<<512, 256, 0, stream>>>(w.B[0], x2, c2g, c2b);

  bottleneck(stream, x1, mA, pkA, w);
  bottleneck(stream, x1, mB, pkB, w);

  // final: silu(BN(conv3(concat(x1, x2)))), f32 NCHW into d_out
  gemm(stream, pc3, x1, (float*)d_out, 128, 128, nullptr, &c3bn, A_SILU,
       nullptr, 1);
}
